// RecNet_PyTorch_24120536334719
// MI455X (gfx1250) — matrix operand order repaired and checked by reading, not compiled, not run on hardware
//
#include <hip/hip_runtime.h>
#include <math.h>

// Problem sizes (fixed by the reference).
#define T_STEPS 8192
#define M_DIM   1024
#define N_DIM   2048

// Recurrent-phase persistent grid: 64 WGs, each owns N/64 = 32 rows of W_hh,
// held resident in LDS (32*2048 bf16 = 128 KB out of 320 KB/WGP).
#define RNN_WGS     64
#define ROWS_PER_WG (N_DIM / RNN_WGS)   // 32 -> 2 row-tiles of 16
#define KSECS       4                   // 8 waves = 2 tiles x 4 K-sections

typedef __attribute__((ext_vector_type(16))) __bf16 v16bf;
typedef __attribute__((ext_vector_type(8)))  __bf16 v8bf;
typedef __attribute__((ext_vector_type(8)))  float  v8f;

typedef __attribute__((ext_vector_type(4))) unsigned int u32x4;
typedef __attribute__((ext_vector_type(8))) int          i32x8;
typedef __attribute__((ext_vector_type(4))) int          i32x4;

static __device__ __forceinline__ __bf16 f2bf(float f) {
  // round-to-nearest-even fp32 -> bf16
  unsigned u = __builtin_bit_cast(unsigned, f);
  unsigned r = (u + 0x7FFFu + ((u >> 16) & 1u)) >> 16;
  unsigned short s = (unsigned short)r;
  return __builtin_bit_cast(__bf16, s);
}

static __device__ __forceinline__ v8f wmma_bf16(v16bf a, v16bf b, v8f c) {
  // (neg_a, A, neg_b, B, c_mod, C, reuse_a, reuse_b)
  return __builtin_amdgcn_wmma_f32_16x16x32_bf16(false, a, false, b, (short)0, c,
                                                 false, false);
}

// A-matrix (16x32 bf16) fragment: per-lane row pointer (row-major source).
// Layout used here (contiguous k, B uses the same): lane = khalf*16 + m; element i -> K = i + khalf*16.
static __device__ __forceinline__ v16bf load_a_frag(const __bf16* rowptr, int kbase,
                                                    int khalf) {
  union { v16bf v; v8bf h[2]; } u;
  const int k0 = kbase + khalf * 16;
  u.h[0] = *(const v8bf*)(rowptr + k0);        // K = k0 .. k0+7
  u.h[1] = *(const v8bf*)(rowptr + k0 + 8);    // K = k0+8 .. k0+15
  return u.v;
}

// ---------------------------------------------------------------------------
// fp32 -> bf16 elementwise convert
__global__ __launch_bounds__(256) void convert_bf16_kernel(const float* __restrict__ src,
                                                           __bf16* __restrict__ dst,
                                                           int n) {
  int i = blockIdx.x * blockDim.x + threadIdx.x;
  int stride = gridDim.x * blockDim.x;
  for (; i < n; i += stride) dst[i] = f2bf(src[i]);
}

// Zero h0 and the grid-barrier counter (d_ws is poisoned each run).
__global__ __launch_bounds__(256) void init_kernel(float* __restrict__ hbuf,
                                                   unsigned* __restrict__ cnt) {
  int i = blockIdx.x * blockDim.x + threadIdx.x;
  if (i == 0) *cnt = 0u;
  for (; i < N_DIM; i += blockDim.x * gridDim.x) hbuf[i] = 0.0f;
}

// ---------------------------------------------------------------------------
// pre[t,n] = sum_m X[t,m]*W_ih[n,m] + (b_ih[n]+b_hh[n]), written to Y (in-place
// consumed by the recurrent kernel). One 16x16 output tile per wave; four
// independent accumulator chains so loads overlap WMMA issue.
__global__ __launch_bounds__(256) void gemm_pre_kernel(const __bf16* __restrict__ Xbf,
                                                       const __bf16* __restrict__ Wihbf,
                                                       const float* __restrict__ bih,
                                                       const float* __restrict__ bhh,
                                                       float* __restrict__ Y) {
  const int wave  = threadIdx.x >> 5;
  const int lane  = threadIdx.x & 31;
  const int khalf = lane >> 4;
  const int l15   = lane & 15;

  const int tileId = blockIdx.x * 8 + wave;     // 65536 tiles total
  const int tn = tileId & 127;                  // N_DIM/16 = 128 tiles
  const int tt = tileId >> 7;                   // T/16 = 512 tiles
  const int tbase = tt * 16;
  const int nbase = tn * 16;

  const __bf16* arow = Xbf   + (size_t)(tbase + l15) * M_DIM;  // X row (row-major)
  const __bf16* bcol = Wihbf + (size_t)(nbase + l15) * M_DIM;  // B col n = W_ih row n

  v8f acc0 = {}, acc1 = {}, acc2 = {}, acc3 = {};
  for (int kb = 0; kb < M_DIM; kb += 128) {
    __builtin_prefetch(arow + kb + 128, 0, 1);  // global_prefetch_b8
    __builtin_prefetch(bcol + kb + 128, 0, 1);
    v16bf a0 = load_a_frag(arow, kb, khalf);
    v16bf b0 = *(const v16bf*)(bcol + kb + khalf * 16);
    acc0 = wmma_bf16(a0, b0, acc0);
    v16bf a1 = load_a_frag(arow, kb + 32, khalf);
    v16bf b1 = *(const v16bf*)(bcol + kb + 32 + khalf * 16);
    acc1 = wmma_bf16(a1, b1, acc1);
    v16bf a2 = load_a_frag(arow, kb + 64, khalf);
    v16bf b2 = *(const v16bf*)(bcol + kb + 64 + khalf * 16);
    acc2 = wmma_bf16(a2, b2, acc2);
    v16bf a3 = load_a_frag(arow, kb + 96, khalf);
    v16bf b3 = *(const v16bf*)(bcol + kb + 96 + khalf * 16);
    acc3 = wmma_bf16(a3, b3, acc3);
  }

  const float bias = bih[nbase + l15] + bhh[nbase + l15];
#pragma unroll
  for (int r = 0; r < 8; ++r) {
    const int row = tbase + r + 8 * khalf;      // C/D layout: m = r + 8*khalf
    Y[(size_t)row * N_DIM + nbase + l15] =
        (acc0[r] + acc1[r]) + (acc2[r] + acc3[r]) + bias;
  }
}

// ---------------------------------------------------------------------------
// Tensor Data Mover copy: contiguous `bytes` from global to LDS as a 1-D tile
// of 8-byte elements. One issue per wave; completion via TENSORcnt.
// 6-arg builtin variant: (u32x4 g0, i32x8 g1, i32x4 g2, i32x4 g3, i32x8 g4, cpol)
static __device__ __forceinline__ void tdm_load_to_lds(const void* gsrc, void* ldst,
                                                       unsigned bytes) {
  const unsigned long long ga = (unsigned long long)gsrc;
  const unsigned lds_addr = (unsigned)(unsigned long long)ldst;  // LDS address = low 32 bits of the pointer
  const unsigned units = bytes >> 3;            // 8-byte elements

  u32x4 g0;
  g0.x = 0x1u;                                  // count=1, is_restore=0, gather off
  g0.y = lds_addr;                              // lds_addr [63:32]
  g0.z = (unsigned)(ga & 0xFFFFFFFFu);          // global_addr [95:64]
  g0.w = (unsigned)((ga >> 32) & 0x1FFFFFFu) | (2u << 30);  // addr[56:32] | type=2

  i32x8 g1;
  g1[0] = (int)(3u << 16);                      // wg_mask=0, data_size=3 (8B), no flags
  g1[1] = (int)(units << 16);                   // barrier_addr=0 | tensor_dim0[15:0]
  g1[2] = (int)((units >> 16) | (1u << 16));    // tensor_dim0[31:16] | tensor_dim1[15:0]=1
  g1[3] = (int)(units << 16);                   // tensor_dim1[31:16]=0 | tile_dim0
  g1[4] = 1;                                    // tile_dim1=1 | tile_dim2=0
  g1[5] = (int)units;                           // tensor_dim0_stride[31:0]
  g1[6] = (int)(units << 16);                   // stride0[47:32]=0 | stride1[15:0]
  g1[7] = (int)(units >> 16);                   // stride1[47:16]

  i32x4 g2 = {0, 0, 0, 0};
  i32x4 g3 = {0, 0, 0, 0};
  i32x8 g4 = {0, 0, 0, 0, 0, 0, 0, 0};
  __builtin_amdgcn_tensor_load_to_lds(g0, g1, g2, g3, g4, 0);
}

// ---------------------------------------------------------------------------
// Persistent recurrent scan. Each WG holds a 32-row slice of W_hh (bf16) in
// LDS for the whole kernel (loaded once via the Tensor Data Mover). Per step:
// broadcast-stage h_{t-1} (bf16) into LDS, matvec via WMMA with h replicated
// across all 16 B-columns, cross-wave reduce in LDS, tanh, publish h_t slice,
// grid barrier.
__global__ __launch_bounds__(256) void rnn_scan_kernel(const __bf16* __restrict__ Whhbf,
                                                       float* __restrict__ hbuf,
                                                       float* __restrict__ Y,
                                                       unsigned* __restrict__ cnt) {
  extern __shared__ char smem[];
  __bf16* Wsl  = (__bf16*)smem;                                     // [32][2048]
  __bf16* hbf  = (__bf16*)(smem + (size_t)ROWS_PER_WG * N_DIM * 2); // [2048]
  float*  part = (float*)(smem + (size_t)ROWS_PER_WG * N_DIM * 2 + N_DIM * 2); // [2][4][16]

  const int tid   = threadIdx.x;
  const int wave  = tid >> 5;
  const int lane  = tid & 31;
  const int khalf = lane >> 4;
  const int l15   = lane & 15;
  const int rowbase = blockIdx.x * ROWS_PER_WG;

  // One-time TDM load of this WG's W_hh slice (contiguous 128 KB) into LDS.
  if (wave == 0) {
    tdm_load_to_lds(Whhbf + (size_t)rowbase * N_DIM, Wsl,
                    (unsigned)(ROWS_PER_WG * N_DIM * 2));
    __builtin_amdgcn_s_wait_tensorcnt(0);
  }

  const int tile = wave & 1;                   // which 16-row tile
  const int ksec = wave >> 1;                  // which 512-wide K section
  const int k0   = ksec * (N_DIM / KSECS);
  const __bf16* arow = Wsl + (size_t)(tile * 16 + l15) * N_DIM;

  for (int t = 0; t < T_STEPS; ++t) {
    // Stage h_{t-1} -> LDS bf16 (visible via prior acquire / init kernel).
    for (int i = tid; i < N_DIM; i += 256) hbf[i] = f2bf(hbuf[i]);
    __syncthreads();   // also publishes the TDM-loaded Wsl on t==0

    v8f acc0 = {}, acc1 = {};
#pragma unroll 4
    for (int kc = 0; kc < N_DIM / KSECS; kc += 64) {
      v16bf a0 = load_a_frag(arow, k0 + kc, khalf);
      v16bf b0 = *(const v16bf*)(hbf + k0 + kc + khalf * 16);   // h replicated to all cols
      acc0 = wmma_bf16(a0, b0, acc0);
      v16bf a1 = load_a_frag(arow, k0 + kc + 32, khalf);
      v16bf b1 = *(const v16bf*)(hbf + k0 + kc + 32 + khalf * 16);
      acc1 = wmma_bf16(a1, b1, acc1);
    }

    // All B-columns are identical; lanes 0 and 16 hold rows 0-7 / 8-15.
    if (l15 == 0) {
#pragma unroll
      for (int r = 0; r < 8; ++r)
        part[(tile * KSECS + ksec) * 16 + khalf * 8 + r] = acc0[r] + acc1[r];
    }
    __syncthreads();

    if (tid < ROWS_PER_WG) {
      const int tl = tid >> 4, rr = tid & 15;
      float s = part[(tl * KSECS + 0) * 16 + rr] + part[(tl * KSECS + 1) * 16 + rr] +
                part[(tl * KSECS + 2) * 16 + rr] + part[(tl * KSECS + 3) * 16 + rr];
      const int grow = rowbase + tid;
      const size_t oidx = (size_t)t * N_DIM + grow;
      const float hv = tanhf(Y[oidx] + s);     // Y currently holds pre (incl. biases)
      Y[oidx] = hv;
      hbuf[grow] = hv;
    }
    __threadfence();
    __syncthreads();

    // Monotonic grid barrier: target = (t+1)*RNN_WGS arrivals.
    if (tid == 0) {
      __hip_atomic_fetch_add(cnt, 1u, __ATOMIC_RELEASE, __HIP_MEMORY_SCOPE_AGENT);
      const unsigned target = (unsigned)(t + 1) * RNN_WGS;
      while (__hip_atomic_load(cnt, __ATOMIC_ACQUIRE, __HIP_MEMORY_SCOPE_AGENT) < target)
        __builtin_amdgcn_s_sleep(1);
    }
    __syncthreads();
  }
}

// ---------------------------------------------------------------------------
extern "C" void kernel_launch(void* const* d_in, const int* in_sizes, int n_in,
                              void* d_out, int out_size, void* d_ws, size_t ws_size,
                              hipStream_t stream) {
  (void)in_sizes; (void)n_in; (void)out_size; (void)ws_size;

  const float* X   = (const float*)d_in[0];   // (T, M)
  const float* Wih = (const float*)d_in[1];   // (N, M)
  const float* Whh = (const float*)d_in[2];   // (N, N)
  const float* bih = (const float*)d_in[3];   // (N,)
  const float* bhh = (const float*)d_in[4];   // (N,)
  float* Y = (float*)d_out;                   // (T, N)

  // Workspace layout (bytes): Xbf 16M | Wihbf 4M | Whhbf 8M | hbuf 8K | cnt
  char* ws = (char*)d_ws;
  __bf16* Xbf   = (__bf16*)ws;
  __bf16* Wihbf = (__bf16*)(ws + (size_t)T_STEPS * M_DIM * 2);
  __bf16* Whhbf = (__bf16*)(ws + (size_t)T_STEPS * M_DIM * 2 + (size_t)N_DIM * M_DIM * 2);
  float*  hbuf  = (float*)(ws + (size_t)T_STEPS * M_DIM * 2 + (size_t)N_DIM * M_DIM * 2 +
                           (size_t)N_DIM * N_DIM * 2);
  unsigned* cnt = (unsigned*)(hbuf + N_DIM);

  convert_bf16_kernel<<<1024, 256, 0, stream>>>(X,   Xbf,   T_STEPS * M_DIM);
  convert_bf16_kernel<<<512,  256, 0, stream>>>(Wih, Wihbf, N_DIM * M_DIM);
  convert_bf16_kernel<<<1024, 256, 0, stream>>>(Whh, Whhbf, N_DIM * N_DIM);
  init_kernel<<<1, 256, 0, stream>>>(hbuf, cnt);

  // 512*128 = 65536 tiles, 8 tiles (waves) per block.
  gemm_pre_kernel<<<8192, 256, 0, stream>>>(Xbf, Wihbf, bih, bhh, Y);

  const size_t smem_bytes = (size_t)ROWS_PER_WG * N_DIM * 2   // W_hh slice
                          + (size_t)N_DIM * 2                 // h bf16
                          + 2 * KSECS * 16 * sizeof(float);   // partials
  rnn_scan_kernel<<<RNN_WGS, 256, smem_bytes, stream>>>(Whhbf, hbuf, Y, cnt);
}
